// MultiResolutionHashTimeDoubleEncoderNetwork_21320217658072
// MI455X (gfx1250) — matrix operand order repaired and checked by reading, not compiled, not run on hardware
//
#include <hip/hip_runtime.h>
#include <cstdint>
#include <cstddef>

// ---------------------------------------------------------------------------
// MultiResolutionHashTimeDoubleEncoderNetwork — fused encode + MLP for gfx1250
//   * hash-grid gathers hit the 192MB L2 (both tables = 128MB resident)
//   * MLP runs on V_WMMA_F32_16X16X32_F16 (wave32), activations staged in LDS
//   * ReLU via single v_med3_f32 (clamp to [0, +inf)) instead of 2x v_max
// ---------------------------------------------------------------------------

typedef _Float16 v16h __attribute__((ext_vector_type(16)));
typedef _Float16 v8h  __attribute__((ext_vector_type(8)));
typedef float    v8f  __attribute__((ext_vector_type(8)));

#define WMMA16x16x32(a, b, c) \
    __builtin_amdgcn_wmma_f32_16x16x32_f16(false, (a), false, (b), (short)0, (c), false, false)

namespace {

constexpr int      L_LVL   = 16;
constexpr uint32_t T_SIZE  = 1u << 19;
constexpr uint32_t T_MASK  = T_SIZE - 1u;
constexpr int      BLK     = 256;       // threads per block = points per block
constexpr int      ACT_S   = 72;        // padded row stride (halves) for act buffer

// floor(16 * (2048/16)^(l/15))
__device__ constexpr int RES_TAB[16] = {
    16, 22, 30, 42, 58, 80, 111, 153, 212, 294, 406, 561, 776, 1072, 1482, 2048};

// LDS layout (in _Float16 units)
constexpr int WT_OFF   = 0;                       // 5 * 64*64 transposed weights
constexpr int W22_OFF  = WT_OFF + 5 * 4096;       // 64-entry head vector
constexpr int FEAT_OFF = W22_OFF + 64;            // 256 x 64 encoded features
constexpr int ACT_OFF  = FEAT_OFF + BLK * 64;     // 8 waves x 16 x ACT_S
constexpr int SMEM_HALVES = ACT_OFF + 8 * 16 * ACT_S;
constexpr size_t SMEM_BYTES = (size_t)SMEM_HALVES * sizeof(_Float16);

// single-instruction ReLU: v_med3_f32(x, 0, +inf)
__device__ __forceinline__ float relu1(float x) {
    return __builtin_amdgcn_fmed3f(x, 0.0f, __builtin_inff());
}

// ---- WMMA fragment loaders (wave32, contiguous k order, same for A and B) -----------

// A (16x32 f16): lane = {m = lid&15, g = lid>>4};
//   halves 0..7  -> K = 32c + 16g + j
//   halves 8..15 -> K = 32c + 16g + 8 + j
__device__ __forceinline__ v16h load_a_frag(const _Float16* base, int stride,
                                            int c, int m, int g) {
    const _Float16* p = base + m * stride + 32 * c + 16 * g;
    const v8h lo = *(const v8h*)(p);
    const v8h hi = *(const v8h*)(p + 8);
    v16h a;
#pragma unroll
    for (int i = 0; i < 8; ++i) { a[i] = lo[i]; a[i + 8] = hi[i]; }
    return a;
}

// B (32x16 f16) from transposed weights Wt[n][k] (stride 64):
//   lane = {n = lid&15, g = lid>>4}; half j -> K = 32c + 16g + j, col n
__device__ __forceinline__ v16h load_b_frag(const _Float16* Wt, int t, int c,
                                            int n, int g) {
    return *(const v16h*)(Wt + (16 * t + n) * 64 + 32 * c + 16 * g);
}

// One 64x64 layer: out = act_fn(in @ W). C/D layout: lane {n,g}, v -> row v+8g.
__device__ __forceinline__ void mlp_layer(const _Float16* inP, int inStride,
                                          const _Float16* Wt,
                                          _Float16* outP, int outStride,
                                          int m, int n, int g,
                                          bool blend, float alpha,
                                          const _Float16* featTile) {
    const v16h a0 = load_a_frag(inP, inStride, 0, m, g);
    const v16h a1 = load_a_frag(inP, inStride, 1, m, g);
#pragma unroll
    for (int t = 0; t < 4; ++t) {
        const v16h b0 = load_b_frag(Wt, t, 0, n, g);
        const v16h b1 = load_b_frag(Wt, t, 1, n, g);
        v8f c = {};
        c = WMMA16x16x32(a0, b0, c);
        c = WMMA16x16x32(a1, b1, c);
#pragma unroll
        for (int v = 0; v < 8; ++v) {
            float h = relu1(c[v]);
            if (blend) {
                const float f0 = (float)featTile[(v + 8 * g) * 64 + 16 * t + n];
                h = alpha * h + (1.0f - alpha) * f0;
            }
            outP[(v + 8 * g) * outStride + 16 * t + n] = (_Float16)h;
        }
    }
}

// Final layer: relu(in @ W) then dot with w22 -> one scalar per point.
__device__ __forceinline__ void head_layer(const _Float16* inP, int inStride,
                                           const _Float16* Wt,
                                           const _Float16* w22,
                                           float* outp, int remaining,
                                           int n, int g) {
    const v16h a0 = load_a_frag(inP, inStride, 0, n, g);
    const v16h a1 = load_a_frag(inP, inStride, 1, n, g);
    float sums[8] = {0.f, 0.f, 0.f, 0.f, 0.f, 0.f, 0.f, 0.f};
#pragma unroll
    for (int t = 0; t < 4; ++t) {
        const v16h b0 = load_b_frag(Wt, t, 0, n, g);
        const v16h b1 = load_b_frag(Wt, t, 1, n, g);
        v8f c = {};
        c = WMMA16x16x32(a0, b0, c);
        c = WMMA16x16x32(a1, b1, c);
        const float wv = (float)w22[16 * t + n];
#pragma unroll
        for (int v = 0; v < 8; ++v)
            sums[v] = fmaf(relu1(c[v]), wv, sums[v]);
    }
    // reduce over the 16 column-lanes of each half-wave
#pragma unroll
    for (int mask = 1; mask <= 8; mask <<= 1) {
#pragma unroll
        for (int v = 0; v < 8; ++v)
            sums[v] += __shfl_xor(sums[v], mask, 32);
    }
    const int row = 8 * g + n;   // lane n (n<8) stores row n + 8g
    if (n < 8 && row < remaining) outp[row] = sums[n];
}

// ---------------------------------------------------------------------------

__global__ __launch_bounds__(BLK)
void hashmlp_fused_kernel(const float* __restrict__ x,
                          const float* __restrict__ t_scalar,
                          const float* __restrict__ tab_static,
                          const float* __restrict__ tab_dynamic,
                          const float* __restrict__ w1_0,
                          const float* __restrict__ w1_1,
                          const float* __restrict__ w1_2,
                          const float* __restrict__ w2_0,
                          const float* __restrict__ w2_1,
                          const float* __restrict__ w2_2,
                          const float* __restrict__ alpha_p,
                          float* __restrict__ out,
                          int npts) {
    extern __shared__ _Float16 smem[];
    _Float16* Wt    = smem + WT_OFF;     // 5 transposed 64x64 matrices
    _Float16* w22   = smem + W22_OFF;
    _Float16* feat0 = smem + FEAT_OFF;   // [256][64]
    _Float16* act   = smem + ACT_OFF;    // [8 waves][16][ACT_S]

    const int tid   = threadIdx.x;
    const float tval  = t_scalar[0];
    const float alpha = alpha_p[0];

    // ---- stage weights into LDS, transposed: Wt[n][k] = W[k][n] ------------
    {
        const float* gw[5] = {w1_0, w1_1, w1_2, w2_0, w2_1};
#pragma unroll
        for (int mi = 0; mi < 5; ++mi) {
            const float* src = gw[mi];
            _Float16* dst = Wt + mi * 4096;
            for (int i = tid; i < 4096; i += BLK) {
                const int nrow = i >> 6, k = i & 63;
                dst[i] = (_Float16)src[k * 64 + nrow];
            }
        }
        if (tid < 64) w22[tid] = (_Float16)w2_2[tid];
    }

    // ---- encode: one point per thread --------------------------------------
    {
        int p = blockIdx.x * BLK + tid;
        if (p >= npts) p = npts - 1;                 // clamp (keeps EXEC full)
        const float xr = x[(size_t)p * 3 + 0];
        const float yr = x[(size_t)p * 3 + 1];
        const float zr = x[(size_t)p * 3 + 2];
        _Float16* frow = feat0 + tid * 64;

        // static: 3-D, 8 corners per level -> cols [0,32)
#pragma unroll
        for (int l = 0; l < L_LVL; ++l) {
            const float res = (float)RES_TAB[l];
            const float* tl = tab_static + (size_t)l * (T_SIZE * 2);
            const float px = xr * res, py = yr * res, pz = zr * res;
            const float gx = floorf(px), gy = floorf(py), gz = floorf(pz);
            const float fx = px - gx, fy = py - gy, fz = pz - gz;
            const uint32_t ix = (uint32_t)(int)gx;
            const uint32_t iy = (uint32_t)(int)gy;
            const uint32_t iz = (uint32_t)(int)gz;
            float a0 = 0.f, a1 = 0.f;
#pragma unroll
            for (int c = 0; c < 8; ++c) {
                const uint32_t h = ((ix + (uint32_t)(c & 1))) ^
                                   ((iy + (uint32_t)((c >> 1) & 1)) * 2654435761u) ^
                                   ((iz + (uint32_t)((c >> 2) & 1)) * 805459861u);
                const float w = ((c & 1) ? fx : 1.f - fx) *
                                (((c >> 1) & 1) ? fy : 1.f - fy) *
                                (((c >> 2) & 1) ? fz : 1.f - fz);
                const float2 f = *(const float2*)(tl + (size_t)(h & T_MASK) * 2);
                a0 = fmaf(w, f.x, a0);
                a1 = fmaf(w, f.y, a1);
            }
            frow[2 * l]     = (_Float16)a0;
            frow[2 * l + 1] = (_Float16)a1;
        }

        // dynamic: 4-D (x,y,z,t), 16 corners per level -> cols [32,64)
#pragma unroll
        for (int l = 0; l < L_LVL; ++l) {
            const float res = (float)RES_TAB[l];
            const float* tl = tab_dynamic + (size_t)l * (T_SIZE * 2);
            const float px = xr * res, py = yr * res, pz = zr * res, pt = tval * res;
            const float gx = floorf(px), gy = floorf(py), gz = floorf(pz), gt = floorf(pt);
            const float fx = px - gx, fy = py - gy, fz = pz - gz, ft = pt - gt;
            const uint32_t ix = (uint32_t)(int)gx;
            const uint32_t iy = (uint32_t)(int)gy;
            const uint32_t iz = (uint32_t)(int)gz;
            const uint32_t it = (uint32_t)(int)gt;
            float a0 = 0.f, a1 = 0.f;
#pragma unroll
            for (int c = 0; c < 16; ++c) {
                const uint32_t h = ((ix + (uint32_t)(c & 1))) ^
                                   ((iy + (uint32_t)((c >> 1) & 1)) * 2654435761u) ^
                                   ((iz + (uint32_t)((c >> 2) & 1)) * 805459861u) ^
                                   ((it + (uint32_t)((c >> 3) & 1)) * 3674653429u);
                const float w = ((c & 1) ? fx : 1.f - fx) *
                                (((c >> 1) & 1) ? fy : 1.f - fy) *
                                (((c >> 2) & 1) ? fz : 1.f - fz) *
                                (((c >> 3) & 1) ? ft : 1.f - ft);
                const float2 f = *(const float2*)(tl + (size_t)(h & T_MASK) * 2);
                a0 = fmaf(w, f.x, a0);
                a1 = fmaf(w, f.y, a1);
            }
            frow[32 + 2 * l]     = (_Float16)a0;
            frow[32 + 2 * l + 1] = (_Float16)a1;
        }
    }

    __syncthreads();

    // ---- MLP: each wave processes two 16-point tiles via WMMA --------------
    const int wave = tid >> 5;
    const int lid  = tid & 31;
    const int n    = lid & 15;     // column lane / A-row lane
    const int g    = lid >> 4;     // half-wave group
    _Float16* actw = act + wave * 16 * ACT_S;

    for (int sub = 0; sub < 2; ++sub) {
        const int rb = wave * 32 + sub * 16;            // tile base (in-block)
        const _Float16* featT = feat0 + rb * 64;
        const int pbase = blockIdx.x * BLK + rb;
        const int remaining = npts - pbase;
        if (remaining <= 0) break;

        // net_1: two hidden + output, ReLU everywhere
        mlp_layer(featT, 64,    Wt + 0 * 4096, actw, ACT_S, n, n, g, false, 0.f, nullptr);
        mlp_layer(actw,  ACT_S, Wt + 1 * 4096, actw, ACT_S, n, n, g, false, 0.f, nullptr);
        // out1 -> adaptive skip blend with feat
        mlp_layer(actw,  ACT_S, Wt + 2 * 4096, actw, ACT_S, n, n, g, true, alpha, featT);
        // net_2 hidden 1
        mlp_layer(actw,  ACT_S, Wt + 3 * 4096, actw, ACT_S, n, n, g, false, 0.f, nullptr);
        // net_2 hidden 2 (ReLU) fused with 64->1 head
        head_layer(actw, ACT_S, Wt + 4 * 4096, w22, out + pbase, remaining, n, g);
    }
}

}  // namespace

extern "C" void kernel_launch(void* const* d_in, const int* in_sizes, int n_in,
                              void* d_out, int out_size, void* d_ws, size_t ws_size,
                              hipStream_t stream) {
    (void)n_in; (void)d_ws; (void)ws_size; (void)in_sizes;
    const float* x     = (const float*)d_in[0];
    const float* t     = (const float*)d_in[1];
    const float* tabS  = (const float*)d_in[2];
    const float* tabD  = (const float*)d_in[3];
    const float* w1_0  = (const float*)d_in[4];
    const float* w1_1  = (const float*)d_in[5];
    const float* w1_2  = (const float*)d_in[6];
    const float* w2_0  = (const float*)d_in[7];
    const float* w2_1  = (const float*)d_in[8];
    const float* w2_2  = (const float*)d_in[9];
    const float* alpha = (const float*)d_in[10];
    float* out = (float*)d_out;

    const int npts   = out_size;                 // [N,1] output
    const int blocks = (npts + BLK - 1) / BLK;

    // 92 KB dynamic LDS (CDNA5 WGP has 320 KB) — opt in above the 64 KB default
    hipFuncSetAttribute((const void*)hashmlp_fused_kernel,
                        hipFuncAttributeMaxDynamicSharedMemorySize,
                        (int)SMEM_BYTES);

    hashmlp_fused_kernel<<<blocks, BLK, SMEM_BYTES, stream>>>(
        x, t, tabS, tabD, w1_0, w1_1, w1_2, w2_0, w2_1, w2_2, alpha, out, npts);
}
